// ParallelBlockPlus_9105330668166
// MI455X (gfx1250) — compile-verified, matrix operand order checked by reading, not run on hardware
//
#include <hip/hip_runtime.h>

// ---------------------------------------------------------------------------
// CoaT ParallelBlock for MI455X (gfx1250, wave32).
// GEMMs: v_wmma_f32_16x16x32_bf16 with async-to-LDS B staging (double buffer).
// ---------------------------------------------------------------------------

typedef __attribute__((ext_vector_type(8)))  unsigned short u16x8;
typedef __attribute__((ext_vector_type(16))) unsigned short u16x16;
typedef __attribute__((ext_vector_type(16))) __bf16         v16bf;
typedef __attribute__((ext_vector_type(8)))  float          v8f;

typedef int v4i_t __attribute__((vector_size(16)));

__device__ __forceinline__ unsigned short f2bf(float f) {
    unsigned u = __builtin_bit_cast(unsigned, f);
    u += 0x7FFFu + ((u >> 16) & 1u);      // round-to-nearest-even
    return (unsigned short)(u >> 16);
}

// one 16-byte async copy global -> LDS (CDNA5 GLOBAL_LOAD_ASYNC_TO_LDS_B128)
__device__ __forceinline__ void async_ld16(const void* g, void* l) {
#if __has_builtin(__builtin_amdgcn_global_load_async_to_lds_b128)
    __builtin_amdgcn_global_load_async_to_lds_b128(
        (__attribute__((address_space(1))) v4i_t*)g,
        (__attribute__((address_space(3))) v4i_t*)l, 0, 0);
#else
    *(u16x16*)l = *(const u16x16*)g;
#endif
}
__device__ __forceinline__ void async_wait_all() {
#if __has_builtin(__builtin_amdgcn_global_load_async_to_lds_b128)
#if __has_builtin(__builtin_amdgcn_s_wait_asynccnt)
    __builtin_amdgcn_s_wait_asynccnt(0);
#else
    asm volatile("s_wait_asynccnt 0x0" ::: "memory");
#endif
#endif
}

// ======================= WMMA GEMM (bf16 -> f32) ===========================
// Block = 8 waves = 128x64 macrotile; per k-step the 4 KB of packed B shared
// by all waves is staged to LDS via async copies (double buffered); A is
// software-pipelined in registers; out-of-range A rows are clamped (stores
// masked), so no exec-mask juggling in the hot loop.
template <bool BIAS, bool GELU_, bool LOADC, bool STOREBF>
__global__ __launch_bounds__(256) void wmma_gemm_kernel(
    const unsigned short* __restrict__ A, long aBatch, int lda,
    const unsigned short* __restrict__ Bp, long bBatch, int nbCount,
    const float* __restrict__ bias,
    const float* Cin,
    float* outF, unsigned short* outB,
    long cOuter, long cInner, int cDiv, int ldc,
    int M, int kBlocks, int nQuads, float alpha)
{
    __shared__ __align__(16) unsigned char bsm[2][4096];

    int tid  = threadIdx.x;
    int wv   = tid >> 5;
    int lane = tid & 31;
    int half = lane >> 4;
    int l15  = lane & 15;

    int nQuad  = blockIdx.x % nQuads;
    int mBlock = blockIdx.x / nQuads;
    int mTile  = mBlock * 8 + wv;

    long z = blockIdx.z;
    const unsigned short* Az = A + z * aBatch;
    const unsigned char*  Bz = (const unsigned char*)(Bp + z * bBatch) + (long)nQuad * 4096;
    long coff = (long)(z / cDiv) * cOuter + (long)(z % cDiv) * cInner;

    int arow = mTile * 16 + l15;
    int rowc = (arow < M) ? arow : (M - 1);
    const unsigned short* ap = Az + (long)rowc * lda + half * 8;
    long bstep = (long)nbCount * 1024;   // bytes per k-step of packed B

    async_ld16(Bz + tid * 16, &bsm[0][tid * 16]);
    async_wait_all();
    __syncthreads();

    u16x8 alo = *(const u16x8*)(ap);
    u16x8 ahi = *(const u16x8*)(ap + 16);

    v8f acc[4];
#pragma unroll
    for (int q = 0; q < 4; ++q)
#pragma unroll
        for (int r = 0; r < 8; ++r) acc[q][r] = 0.0f;

    for (int kb = 0; kb < kBlocks; ++kb) {
        int nxt = kb + 1;
        if (nxt < kBlocks)
            async_ld16(Bz + (long)nxt * bstep + tid * 16, &bsm[nxt & 1][tid * 16]);
        u16x8 nlo = alo, nhi = ahi;
        if (nxt < kBlocks) {
            nlo = *(const u16x8*)(ap + nxt * 32);
            nhi = *(const u16x8*)(ap + nxt * 32 + 16);
        }

        u16x16 au;
#pragma unroll
        for (int t = 0; t < 8; ++t) { au[t] = alo[t]; au[8 + t] = ahi[t]; }
        v16bf afr = __builtin_bit_cast(v16bf, au);

        const unsigned short* bl = (const unsigned short*)bsm[kb & 1];
#pragma unroll
        for (int q = 0; q < 4; ++q) {
            u16x16 bu = *(const u16x16*)(bl + q * 512 + lane * 16);   // ds_load_b128 x2
            v16bf bfr = __builtin_bit_cast(v16bf, bu);
            acc[q] = __builtin_amdgcn_wmma_f32_16x16x32_bf16(
                false, afr, false, bfr, (short)0, acc[q], false, false);
        }

        if (nxt < kBlocks) {
            async_wait_all();
            __syncthreads();
        }
        alo = nlo; ahi = nhi;
    }

#pragma unroll
    for (int q = 0; q < 4; ++q) {
        int col = nQuad * 64 + q * 16 + l15;
        float bc = BIAS ? bias[col] : 0.0f;
#pragma unroll
        for (int r = 0; r < 8; ++r) {
            int row = mTile * 16 + r + half * 8;
            if (row < M) {
                float v = alpha * acc[q][r];
                if (BIAS)  v += bc;
                if (GELU_) v = 0.5f * v * (1.0f + erff(v * 0.70710678118654752f));
                long o = coff + (long)row * ldc + col;
                if (LOADC) v += Cin[o];
                if (STOREBF) outB[o] = f2bf(v);
                else         outF[o] = v;
            }
        }
    }
}

// ================= pack f32 source -> bf16 B-fragment blocks ===============
__global__ void pack_b_kernel(const float* __restrict__ src,
                              long zOuter, long zInner, int zDiv, long colOff,
                              int ldsrc, int kbCount, int nbCount, int Krows,
                              unsigned short* __restrict__ dst, long total)
{
    long idx = (long)blockIdx.x * blockDim.x + threadIdx.x;
    if (idx >= total) return;
    int  e    = (int)(idx & 15);
    int  lane = (int)((idx >> 4) & 31);
    long blk  = idx >> 9;
    int  nb   = (int)(blk % nbCount);
    long r2   = blk / nbCount;
    int  kb   = (int)(r2 % kbCount);
    long z    = r2 / kbCount;
    int  half = lane >> 4;
    int  kin  = e + ((e >= 8) ? 8 : 0) + 8 * half;
    int  gk   = kb * 32 + kin;
    int  n    = nb * 16 + (lane & 15);
    long sb   = (z / zDiv) * zOuter + (z % zDiv) * zInner + colOff;
    float v = (gk < Krows) ? src[sb + (long)gk * ldsrc + n] : 0.0f;
    dst[idx] = f2bf(v);
}

// ======================= CPE: dwconv3x3 + residual =========================
__global__ void cpe_kernel(const float* __restrict__ x, const float* __restrict__ w,
                           const float* __restrict__ bias, float* __restrict__ xp,
                           int H, int W, int N, long total)
{
    long idx = (long)blockIdx.x * blockDim.x + threadIdx.x;
    if (idx >= total) return;
    int  c = (int)(idx & 511);
    long r = idx >> 9;
    int  n = (int)(r % N);
    int  b = (int)(r / N);
    if (n == 0) { xp[idx] = x[idx]; return; }
    int pix = n - 1, hh = pix / W, ww = pix % W;
    float s = 0.0f;
#pragma unroll
    for (int dy = 0; dy < 3; ++dy) {
        int yy = hh + dy - 1;
        if (yy < 0 || yy >= H) continue;
#pragma unroll
        for (int dx = 0; dx < 3; ++dx) {
            int xx = ww + dx - 1;
            if (xx < 0 || xx >= W) continue;
            s += w[c * 9 + dy * 3 + dx] * x[((long)b * N + 1 + yy * W + xx) * 512 + c];
        }
    }
    xp[idx] = s + bias[c] + x[idx];
}

// ===================== LayerNorm (C=512) -> bf16 ===========================
__global__ __launch_bounds__(128) void ln_kernel(const float* __restrict__ x,
                                                 const float* __restrict__ g,
                                                 const float* __restrict__ bb,
                                                 unsigned short* __restrict__ out)
{
    long row = blockIdx.x;
    int  t = threadIdx.x;
    const float* xr = x + row * 512;
    float4 v = ((const float4*)xr)[t];
    __shared__ float s1[128], s2[128];
    s1[t] = v.x + v.y + v.z + v.w;
    s2[t] = v.x * v.x + v.y * v.y + v.z * v.z + v.w * v.w;
    __syncthreads();
    for (int o = 64; o > 0; o >>= 1) {
        if (t < o) { s1[t] += s1[t + o]; s2[t] += s2[t + o]; }
        __syncthreads();
    }
    float mean = s1[0] * (1.0f / 512.0f);
    float var  = s2[0] * (1.0f / 512.0f) - mean * mean;
    float rstd = rsqrtf(var + 1e-5f);
    int c = t * 4;
    unsigned short* orow = out + row * 512;
    orow[c + 0] = f2bf((v.x - mean) * rstd * g[c + 0] + bb[c + 0]);
    orow[c + 1] = f2bf((v.y - mean) * rstd * g[c + 1] + bb[c + 1]);
    orow[c + 2] = f2bf((v.z - mean) * rstd * g[c + 2] + bb[c + 2]);
    orow[c + 3] = f2bf((v.w - mean) * rstd * g[c + 3] + bb[c + 3]);
}

// ============ softmax over tokens for k, stored transposed bf16 ============
__global__ __launch_bounds__(256) void softmax_k_kernel(const float* __restrict__ qkv,
                                                        unsigned short* __restrict__ ksm,
                                                        int N, int Npad)
{
    int z = blockIdx.x, b = z >> 3, h = z & 7;
    int t = threadIdx.x, ch = t & 63, seg = t >> 6;
    const float* kp = qkv + ((long)b * N) * 1536 + 512 + h * 64 + ch;
    __shared__ float red[4][64];
    float mx = -3.4e38f;
    for (int n = seg; n < N; n += 4) mx = fmaxf(mx, kp[(long)n * 1536]);
    red[seg][ch] = mx;
    __syncthreads();
    mx = fmaxf(fmaxf(red[0][ch], red[1][ch]), fmaxf(red[2][ch], red[3][ch]));
    __syncthreads();
    float sm = 0.0f;
    for (int n = seg; n < N; n += 4) sm += expf(kp[(long)n * 1536] - mx);
    red[seg][ch] = sm;
    __syncthreads();
    sm = red[0][ch] + red[1][ch] + red[2][ch] + red[3][ch];
    float inv = 1.0f / sm;
    unsigned short* op = ksm + ((long)z * 64 + ch) * Npad;
    for (int n = seg; n < N; n += 4) op[n] = f2bf(expf(kp[(long)n * 1536] - mx) * inv);
    for (int n = N + seg; n < Npad; n += 4) op[n] = 0;
}

// ================= gather q -> bf16 [z=(b,h)][n][64] =======================
__global__ void qpack_kernel(const float* __restrict__ qkv, unsigned short* __restrict__ qbf,
                             int N, long total)
{
    long idx = (long)blockIdx.x * blockDim.x + threadIdx.x;
    if (idx >= total) return;
    int  ch = (int)(idx & 63);
    long r  = idx >> 6;
    int  n  = (int)(r % N);
    long z  = r / N;
    int  b  = (int)(z >> 3), h = (int)(z & 7);
    qbf[idx] = f2bf(qkv[((long)b * N + n) * 1536 + h * 64 + ch]);
}

// ============================ f32 -> bf16 ==================================
__global__ void conv_bf_kernel(const float* __restrict__ s, unsigned short* __restrict__ d, long total)
{
    long idx = (long)blockIdx.x * blockDim.x + threadIdx.x;
    if (idx >= total) return;
    d[idx] = f2bf(s[idx]);
}

// ================= CRPE: dwconv on v, multiply by q ========================
__global__ void crpe_kernel(const float* __restrict__ qkv, float* __restrict__ obuf,
                            const float* __restrict__ w3, const float* __restrict__ b3,
                            const float* __restrict__ w5, const float* __restrict__ b5,
                            const float* __restrict__ w7, const float* __restrict__ b7,
                            int H, int W, int N, long total)
{
    long idx = (long)blockIdx.x * blockDim.x + threadIdx.x;
    if (idx >= total) return;
    int  cc  = (int)(idx & 511);
    long p   = idx >> 9;
    int  pix = (int)(p % (H * W));
    int  b   = (int)(p / (H * W));
    int  hh  = pix / W, ww = pix % W;
    int  head = cc >> 6;
    const float* wt; const float* bt; int ks, cidx;
    if (head < 2)      { wt = w3; bt = b3; ks = 3; cidx = cc; }
    else if (head < 5) { wt = w5; bt = b5; ks = 5; cidx = cc - 128; }
    else               { wt = w7; bt = b7; ks = 7; cidx = cc - 320; }
    int r = ks >> 1;
    const float* vb = qkv + ((long)b * N) * 1536 + 1024 + cc;
    float s = 0.0f;
    for (int dy = 0; dy < ks; ++dy) {
        int yy = hh + dy - r;
        if (yy < 0 || yy >= H) continue;
        for (int dx = 0; dx < ks; ++dx) {
            int xx = ww + dx - r;
            if (xx < 0 || xx >= W) continue;
            s += wt[(long)cidx * ks * ks + dy * ks + dx] * vb[(long)(1 + yy * W + xx) * 1536];
        }
    }
    float qv = qkv[((long)b * N + 1 + pix) * 1536 + cc];
    obuf[((long)b * N + 1 + pix) * 512 + cc] = qv * (s + bt[cidx]);
}

__global__ void zerocls_kernel(float* __restrict__ obuf, long rowStride)
{
    int idx = blockIdx.x * blockDim.x + threadIdx.x;
    if (idx >= 16 * 512) return;
    obuf[(long)(idx >> 9) * rowStride + (idx & 511)] = 0.0f;
}

// ============== cross-scale combine with antialiased bilinear ==============
__device__ float resize_sample(const float* __restrict__ cur, int Nj, int Hj, int Wj,
                               int Hi, int Wi, int b, int hi, int wi, int c)
{
    float scy = (float)Hj / Hi, scx = (float)Wj / Wi;
    float ssy = scy > 1.0f ? scy : 1.0f, ssx = scx > 1.0f ? scx : 1.0f;
    float sy = (hi + 0.5f) * scy - 0.5f, sx = (wi + 0.5f) * scx - 0.5f;
    int y0 = (int)ceilf(sy - ssy), y1 = (int)floorf(sy + ssy);
    int x0 = (int)ceilf(sx - ssx), x1 = (int)floorf(sx + ssx);
    float vs = 0.0f, wsum = 0.0f;
    for (int y = y0; y <= y1; ++y) {
        if (y < 0 || y >= Hj) continue;
        float wy = 1.0f - fabsf((float)y - sy) / ssy;
        if (wy <= 0.0f) continue;
        for (int x = x0; x <= x1; ++x) {
            if (x < 0 || x >= Wj) continue;
            float wx = 1.0f - fabsf((float)x - sx) / ssx;
            if (wx <= 0.0f) continue;
            float w = wy * wx;
            vs += w * cur[((long)b * Nj + 1 + y * Wj + x) * 512 + c];
            wsum += w;
        }
    }
    return vs / wsum;
}

__global__ void combine_kernel(const float* __restrict__ feat, const float* __restrict__ ci,
                               const float* __restrict__ ca, int Ha, int Wa, int Na,
                               const float* __restrict__ cb, int Hb, int Wb, int Nb,
                               int Hi, int Wi, int Ni, float* __restrict__ xbuf, long total)
{
    long idx = (long)blockIdx.x * blockDim.x + threadIdx.x;
    if (idx >= total) return;
    int  c = (int)(idx & 511);
    long r = idx >> 9;
    int  n = (int)(r % Ni);
    int  b = (int)(r / Ni);
    float v = feat[idx] + ci[idx];
    if (n == 0) {
        v += ca[((long)b * Na) * 512 + c] + cb[((long)b * Nb) * 512 + c];
    } else {
        int pix = n - 1, hi = pix / Wi, wi = pix % Wi;
        v += resize_sample(ca, Na, Ha, Wa, Hi, Wi, b, hi, wi, c);
        v += resize_sample(cb, Nb, Hb, Wb, Hi, Wi, b, hi, wi, c);
    }
    xbuf[idx] = v;
}

// ============================== host side ==================================
static inline long cdivl(long a, long b) { return (a + b - 1) / b; }

constexpr long O_XP  = 0;
constexpr long O_LN  = O_XP  + 50192L * 512 * 4;
constexpr long O_QKV = O_LN  + 50192L * 512 * 2;
constexpr long O_KSM = O_QKV + 50192L * 1536 * 4;
constexpr long O_VP  = O_KSM + 128L * 64 * 3168 * 2;
constexpr long O_QB  = O_VP  + 128L * 3168 * 64 * 2;
constexpr long O_KVB = O_QB  + 128L * 3137 * 64 * 2;
constexpr long O_KVP = O_KVB + 128L * 4096 * 4;
constexpr long O_OB  = O_KVP + 128L * 4096 * 2;
constexpr long O_C0  = O_OB  + 50192L * 512 * 4;
constexpr long O_C1  = O_C0  + 50192L * 512 * 4;
constexpr long O_C2  = O_C1  + 12560L * 512 * 4;
constexpr long O_WQ  = O_C2  + 3152L  * 512 * 4;
constexpr long O_WP  = O_WQ  + 512L * 1536 * 2;
constexpr long O_WF1 = O_WP  + 512L * 512 * 2;
constexpr long O_WF2 = O_WF1 + 512L * 2048 * 2;

extern "C" void kernel_launch(void* const* d_in, const int* in_sizes, int n_in,
                              void* d_out, int out_size, void* d_ws, size_t ws_size,
                              hipStream_t stream)
{
    (void)in_sizes; (void)n_in; (void)out_size; (void)ws_size;
    const int Hs[3] = {56, 28, 14}, Ws[3] = {56, 28, 14};
    const int Ns[3] = {3137, 785, 197}, Npads[3] = {3168, 800, 224};
    const int Ms[3] = {50192, 12560, 3152};
    const long outOff[3] = {0L, 50192L * 512, 50192L * 512 + 12560L * 512};
    const float SCALE = 0.125f;

    const float* x_in[3] = {(const float*)d_in[0], (const float*)d_in[1], (const float*)d_in[2]};
    const float* cpe_w  = (const float*)d_in[3];
    const float* cpe_b  = (const float*)d_in[4];
    const float* n1_g   = (const float*)d_in[5];
    const float* n1_b   = (const float*)d_in[6];
    const float* n2_g   = (const float*)d_in[7];
    const float* n2_b   = (const float*)d_in[8];
    const float* qkv_w  = (const float*)d_in[9];
    const float* proj_w = (const float*)d_in[10];
    const float* proj_b = (const float*)d_in[11];
    const float* cr3_w  = (const float*)d_in[12];
    const float* cr3_b  = (const float*)d_in[13];
    const float* cr5_w  = (const float*)d_in[14];
    const float* cr5_b  = (const float*)d_in[15];
    const float* cr7_w  = (const float*)d_in[16];
    const float* cr7_b  = (const float*)d_in[17];
    const float* fc1_w  = (const float*)d_in[18];
    const float* fc1_b  = (const float*)d_in[19];
    const float* fc2_w  = (const float*)d_in[20];
    const float* fc2_b  = (const float*)d_in[21];

    char* ws = (char*)d_ws;
    float*          xp    = (float*)(ws + O_XP);
    unsigned short* lnbf  = (unsigned short*)(ws + O_LN);
    float*          qkvB  = (float*)(ws + O_QKV);
    unsigned short* hidden= (unsigned short*)(ws + O_QKV);
    unsigned short* ksm   = (unsigned short*)(ws + O_KSM);
    unsigned short* obf   = (unsigned short*)(ws + O_KSM);
    unsigned short* vpk   = (unsigned short*)(ws + O_VP);
    unsigned short* qbf   = (unsigned short*)(ws + O_QB);
    float*          kvbuf = (float*)(ws + O_KVB);
    unsigned short* kvpk  = (unsigned short*)(ws + O_KVP);
    float*          obuf  = (float*)(ws + O_OB);
    float*          cur[3]= {(float*)(ws + O_C0), (float*)(ws + O_C1), (float*)(ws + O_C2)};
    unsigned short* qkvp  = (unsigned short*)(ws + O_WQ);
    unsigned short* projp = (unsigned short*)(ws + O_WP);
    unsigned short* fc1p  = (unsigned short*)(ws + O_WF1);
    unsigned short* fc2p  = (unsigned short*)(ws + O_WF2);
    float*          dout  = (float*)d_out;

    const int BS = 256;

    {
        long tot = 1L * 16 * 128 * 512;
        pack_b_kernel<<<dim3((unsigned)cdivl(tot, BS)), BS, 0, stream>>>(
            fc1_w, 0, 0, 1, 0, 2048, 16, 128, 512, fc1p, tot);
        tot = 1L * 64 * 32 * 512;
        pack_b_kernel<<<dim3((unsigned)cdivl(tot, BS)), BS, 0, stream>>>(
            fc2_w, 0, 0, 1, 0, 512, 64, 32, 512, fc2p, tot);
    }

    for (int i = 0; i < 3; ++i) {
        const int H = Hs[i], W = Ws[i], N = Ns[i], Npad = Npads[i], M = Ms[i];
        const long MC = (long)M * 512;
        const int mB = (int)cdivl(M, 128);

        cpe_kernel<<<dim3((unsigned)cdivl(MC, BS)), BS, 0, stream>>>(
            x_in[i], cpe_w + (long)i * 512 * 9, cpe_b + i * 512, xp, H, W, N, MC);

        ln_kernel<<<dim3((unsigned)M), 128, 0, stream>>>(xp, n1_g + i * 512, n1_b + i * 512, lnbf);

        {
            long tot = 1L * 16 * 96 * 512;
            pack_b_kernel<<<dim3((unsigned)cdivl(tot, BS)), BS, 0, stream>>>(
                qkv_w + (long)i * 512 * 1536, 0, 0, 1, 0, 1536, 16, 96, 512, qkvp, tot);
        }
        wmma_gemm_kernel<false, false, false, false>
            <<<dim3((unsigned)(mB * 24), 1, 1), 256, 0, stream>>>(
            lnbf, 0, 512, qkvp, 0, 96, nullptr, nullptr, qkvB, nullptr,
            0, 0, 1, 1536, M, 16, 24, 1.0f);

        softmax_k_kernel<<<dim3(128), 256, 0, stream>>>(qkvB, ksm, N, Npad);

        {
            long tot = 128L * (Npad / 32) * 4 * 512;
            pack_b_kernel<<<dim3((unsigned)cdivl(tot, BS)), BS, 0, stream>>>(
                qkvB, (long)N * 1536, 64, 8, 1024, 1536, Npad / 32, 4, N, vpk, tot);
        }

        {
            long tot = 128L * N * 64;
            qpack_kernel<<<dim3((unsigned)cdivl(tot, BS)), BS, 0, stream>>>(qkvB, qbf, N, tot);
        }

        wmma_gemm_kernel<false, false, false, false>
            <<<dim3(1, 1, 128), 256, 0, stream>>>(
            ksm, 64L * Npad, Npad, vpk, (long)Npad * 64, 4, nullptr, nullptr, kvbuf, nullptr,
            4096L, 0, 1, 64, 64, Npad / 32, 1, 1.0f);

        {
            long tot = 128L * 2 * 4 * 512;
            pack_b_kernel<<<dim3((unsigned)cdivl(tot, BS)), BS, 0, stream>>>(
                kvbuf, 4096, 0, 1, 0, 64, 2, 4, 64, kvpk, tot);
        }

        {
            long tot = (long)16 * H * W * 512;
            crpe_kernel<<<dim3((unsigned)cdivl(tot, BS)), BS, 0, stream>>>(
                qkvB, obuf,
                cr3_w + (long)i * 128 * 9,  cr3_b + i * 128,
                cr5_w + (long)i * 192 * 25, cr5_b + i * 192,
                cr7_w + (long)i * 192 * 49, cr7_b + i * 192, H, W, N, tot);
            zerocls_kernel<<<dim3(32), 256, 0, stream>>>(obuf, (long)N * 512);
        }

        {
            int mBf = (int)cdivl(N, 128);
            wmma_gemm_kernel<false, false, true, false>
                <<<dim3((unsigned)mBf, 1, 128), 256, 0, stream>>>(
                qbf, (long)N * 64, 64, kvpk, 4096, 4, nullptr, obuf, obuf, nullptr,
                (long)N * 512, 64, 8, 512, N, 2, 1, SCALE);
        }

        conv_bf_kernel<<<dim3((unsigned)cdivl(MC, BS)), BS, 0, stream>>>(obuf, obf, MC);
        {
            long tot = 1L * 16 * 32 * 512;
            pack_b_kernel<<<dim3((unsigned)cdivl(tot, BS)), BS, 0, stream>>>(
                proj_w + (long)i * 512 * 512, 0, 0, 1, 0, 512, 16, 32, 512, projp, tot);
        }
        wmma_gemm_kernel<true, false, false, false>
            <<<dim3((unsigned)(mB * 8), 1, 1), 256, 0, stream>>>(
            obf, 0, 512, projp, 0, 32, proj_b + i * 512, nullptr, cur[i], nullptr,
            0, 0, 1, 512, M, 16, 8, 1.0f);
    }

    for (int i = 0; i < 3; ++i) {
        const int N = Ns[i], M = Ms[i];
        const long MC = (long)M * 512;
        const int mB = (int)cdivl(M, 128);
        int a = (i == 0) ? 1 : 0;
        int b = (i == 2) ? 1 : 2;

        combine_kernel<<<dim3((unsigned)cdivl(MC, BS)), BS, 0, stream>>>(
            x_in[i], cur[i],
            cur[a], Hs[a], Ws[a], Ns[a],
            cur[b], Hs[b], Ws[b], Ns[b],
            Hs[i], Ws[i], N, xp, MC);

        ln_kernel<<<dim3((unsigned)M), 128, 0, stream>>>(xp, n2_g + i * 512, n2_b + i * 512, lnbf);

        wmma_gemm_kernel<true, true, false, true>
            <<<dim3((unsigned)(mB * 32), 1, 1), 256, 0, stream>>>(
            lnbf, 0, 512, fc1p, 0, 128, fc1_b, nullptr, nullptr, hidden,
            0, 0, 1, 2048, M, 16, 32, 1.0f);

        wmma_gemm_kernel<true, false, true, false>
            <<<dim3((unsigned)(mB * 8), 1, 1), 256, 0, stream>>>(
            hidden, 0, 2048, fc2p, 0, 32, fc2_b, xp, dout + outOff[i], nullptr,
            0, 0, 1, 512, M, 64, 8, 1.0f);
    }
}
